// ContrastiveLoss_87729001988362
// MI455X (gfx1250) — matrix operand order repaired and checked by reading, not compiled, not run on hardware
//
#include <hip/hip_runtime.h>

typedef __attribute__((__ext_vector_type__(16))) __bf16        v16bf;
typedef __attribute__((__ext_vector_type__(8)))  float         v8f;
typedef __attribute__((__ext_vector_type__(4)))  unsigned int  u32x4;
typedef __attribute__((__ext_vector_type__(2)))  unsigned int  u32x2;
typedef __attribute__((__ext_vector_type__(4)))  float         f32x4;

#define NBATCH   4096
#define DIM      1024
#define NROW     8192        // 2*NBATCH
#define TEMP_INV 2.0f        // 1/0.5

__device__ __forceinline__ unsigned short f32_to_bf16(float f) {
  unsigned int x = __float_as_uint(f);
  x += 0x7fffu + ((x >> 16) & 1u);          // round-to-nearest-even
  return (unsigned short)(x >> 16);
}

// ---------------------------------------------------------------------------
// Kernel 0: zero the scalar output (d_out is poisoned by the harness)
// ---------------------------------------------------------------------------
__global__ void nt_xent_zero(float* out) { out[0] = 0.0f; }

// ---------------------------------------------------------------------------
// Kernel 1: row-wise L2 normalize fp32 -> bf16 reps[8192][1024]
// one block per row, 256 threads, 4 elements/thread
// ---------------------------------------------------------------------------
__global__ void nt_xent_normalize(const float* __restrict__ ei,
                                  const float* __restrict__ ej,
                                  unsigned short* __restrict__ reps) {
  const int r = blockIdx.x;
  const float* src = (r < NBATCH) ? (ei + (size_t)r * DIM)
                                  : (ej + (size_t)(r - NBATCH) * DIM);
  const int t = threadIdx.x;
  f32x4 v = *(const f32x4*)(src + t * 4);
  float s = v.x * v.x + v.y * v.y + v.z * v.z + v.w * v.w;
  #pragma unroll
  for (int o = 16; o >= 1; o >>= 1) s += __shfl_xor(s, o, 32);
  __shared__ float wsum[8];
  if ((t & 31) == 0) wsum[t >> 5] = s;
  __syncthreads();
  float tot = 0.0f;
  #pragma unroll
  for (int i = 0; i < 8; ++i) tot += wsum[i];
  const float inv = 1.0f / fmaxf(sqrtf(tot), 1e-12f);

  u32x2 p;
  p.x = (unsigned int)f32_to_bf16(v.x * inv) |
        ((unsigned int)f32_to_bf16(v.y * inv) << 16);
  p.y = (unsigned int)f32_to_bf16(v.z * inv) |
        ((unsigned int)f32_to_bf16(v.w * inv) << 16);
  *(u32x2*)(reps + (size_t)r * DIM + t * 4) = p;
}

// ---------------------------------------------------------------------------
// Kernel 2: fused sim = reps @ reps^T  (bf16 WMMA, fp32 acc) + NT-Xent reduce.
// grid = 128 blocks x 64 rows; 256 threads = 8 waves.
// A-stripe (64x1024 bf16 = 128KB) staged in LDS once; each wave owns a
// disjoint 64-column stripe per iteration (4 B frags -> 16 WMMA per K-chunk,
// halving LDS bytes per WMMA vs a 32-col tile), B streamed from L2.
// No barriers in the main loop.
// ---------------------------------------------------------------------------
__global__ void __launch_bounds__(256, 1)
nt_xent_main(const unsigned short* __restrict__ reps, float* __restrict__ out) {
  extern __shared__ char smem[];
  unsigned short* Alds   = (unsigned short*)smem;               // 64*1024 bf16
  float*          rowsum = (float*)(smem + 64 * DIM * 2);       // [64]
  float*          poslds = rowsum + 64;                         // [64]
  float*          total  = poslds + 64;                         // [1]

  const int t    = threadIdx.x;
  const int w    = t >> 5;         // wave id 0..7
  const int lane = t & 31;
  const int nl   = lane & 15;      // N / M position within 16
  const int hi   = lane >> 4;      // half-wave selector
  const int R0   = blockIdx.x * 64;

  // ---- stage A stripe: 64 rows x 1024 bf16 (128KB), coalesced b128 loads
  {
    const u32x4* src = (const u32x4*)(reps + (size_t)R0 * DIM);
    u32x4*       dst = (u32x4*)Alds;
    #pragma unroll
    for (int i = 0; i < 32; ++i) dst[i * 256 + t] = src[i * 256 + t];
  }
  if (t < 64) { rowsum[t] = 0.0f; poslds[t] = 0.0f; }
  if (t == 0) total[0] = 0.0f;
  __syncthreads();

  float dsum[4][8];
  #pragma unroll
  for (int rt = 0; rt < 4; ++rt)
    #pragma unroll
    for (int i = 0; i < 8; ++i) dsum[rt][i] = 0.0f;

  union Frag { v16bf v; u32x4 q[2]; };

  // ---- 16 column iterations: wave covers cols [it*512 + w*64, +64)
  for (int it = 0; it < 16; ++it) {
    const int cbase = it * 512 + w * 64;

    v8f acc[4][4];
    #pragma unroll
    for (int rt = 0; rt < 4; ++rt)
      #pragma unroll
      for (int bt = 0; bt < 4; ++bt) acc[rt][bt] = (v8f)(0.0f);

    // K loop: 1024 in chunks of 32 (bf16 WMMA K)
    #pragma unroll 2
    for (int kc = 0; kc < 32; ++kc) {
      const int kb = kc * 32;

      // A fragments (LDS): lane holds row M=nl, K = kb+hi*8..+7 and kb+16+hi*8..+7
      Frag a[4];
      #pragma unroll
      for (int rt = 0; rt < 4; ++rt) {
        const unsigned short* ap = Alds + (size_t)(rt * 16 + nl) * DIM + kb + hi * 8;
        a[rt].q[0] = *(const u32x4*)ap;         // K = kb + hi*8      .. +7
        a[rt].q[1] = *(const u32x4*)(ap + 16);  // K = kb + 16 + hi*8 .. +7
      }
      // B fragments (global/L2): lane holds col N=nl, K = kb+hi*8..+7 and kb+16+hi*8..+7
      Frag b[4];
      #pragma unroll
      for (int bt = 0; bt < 4; ++bt) {
        const unsigned short* bp =
            reps + (size_t)(cbase + bt * 16 + nl) * DIM + kb + hi * 8;
        b[bt].q[0] = *(const u32x4*)bp;
        b[bt].q[1] = *(const u32x4*)(bp + 16);
      }
      #pragma unroll
      for (int rt = 0; rt < 4; ++rt)
        #pragma unroll
        for (int bt = 0; bt < 4; ++bt)
          acc[rt][bt] = __builtin_amdgcn_wmma_f32_16x16x32_bf16(
              false, a[rt].v, false, b[bt].v, (short)0, acc[rt][bt],
              false, false);
    }

    // ---- epilogue: exp(2*s), mask diagonal, capture positives, row sums
    #pragma unroll
    for (int rt = 0; rt < 4; ++rt) {
      #pragma unroll
      for (int i = 0; i < 8; ++i) {
        const int rloc = rt * 16 + i + hi * 8;   // C layout: M = i + hi*8
        const int rg   = R0 + rloc;
        #pragma unroll
        for (int bt = 0; bt < 4; ++bt) {
          const float s = acc[rt][bt][i];
          const int   c = cbase + bt * 16 + nl;  // C layout: N = nl
          const float e = __expf(TEMP_INV * s);
          dsum[rt][i] += (c == rg) ? 0.0f : e;   // neg_mask removes diagonal only
          if (c == (rg ^ NBATCH)) poslds[rloc] = s;  // partner = r XOR 4096
        }
      }
    }
  }

  // ---- combine per-lane partial row sums (ds_add_f32)
  #pragma unroll
  for (int rt = 0; rt < 4; ++rt)
    #pragma unroll
    for (int i = 0; i < 8; ++i)
      atomicAdd(&rowsum[rt * 16 + i + hi * 8], dsum[rt][i]);
  __syncthreads();

  if (t < 64) {
    const float v = __logf(rowsum[t]) - TEMP_INV * poslds[t];
    atomicAdd(total, v);
  }
  __syncthreads();
  if (t == 0) atomicAdd(out, total[0] * (1.0f / (float)NROW));
}

// ---------------------------------------------------------------------------
extern "C" void kernel_launch(void* const* d_in, const int* in_sizes, int n_in,
                              void* d_out, int out_size, void* d_ws, size_t ws_size,
                              hipStream_t stream) {
  (void)in_sizes; (void)n_in; (void)out_size; (void)ws_size;
  const float* ei = (const float*)d_in[0];
  const float* ej = (const float*)d_in[1];
  float* out = (float*)d_out;
  unsigned short* reps = (unsigned short*)d_ws;   // 8192*1024 bf16 = 16.8MB

  const size_t smem = (size_t)64 * DIM * 2 + (64 + 64 + 1) * sizeof(float);
  (void)hipFuncSetAttribute((const void*)nt_xent_main,
                            hipFuncAttributeMaxDynamicSharedMemorySize,
                            (int)smem);

  nt_xent_zero<<<1, 1, 0, stream>>>(out);
  nt_xent_normalize<<<NROW, 256, 0, stream>>>(ei, ej, reps);
  nt_xent_main<<<NROW / 64, 256, smem, stream>>>(reps, out);
}
